// Model_41893111005331
// MI455X (gfx1250) — matrix operand order repaired and checked by reading, not compiled, not run on hardware
//
#include <hip/hip_runtime.h>
#include <math.h>

// MI455X / gfx1250, wave32 only.
typedef __attribute__((ext_vector_type(16))) _Float16 v16h;
typedef __attribute__((ext_vector_type(8)))  float    v8f;
typedef __attribute__((ext_vector_type(4)))  unsigned int u32x4;
typedef __attribute__((ext_vector_type(8)))  int      i32x8;
typedef __attribute__((ext_vector_type(4)))  int      i32x4;

#if defined(__has_builtin)
# if __has_builtin(__builtin_amdgcn_tensor_load_to_lds)
#  define HAVE_TDM 1
# endif
#endif
#ifndef HAVE_TDM
# define HAVE_TDM 0
#endif

// Problem constants (from reference): B=2, N=64, T=64, D=128, E=4, H=32, BT=128
constexpr int SZ = 136;  // f16 row stride for 128-wide tiles (pad vs 64-bank LDS)
constexpr int SA = 72;   // f16 row stride for 64-wide alpha
constexpr int SL = 68;   // f32 row stride for 64-wide logits
constexpr int SO = 132;  // f32 row stride for 128-wide f32 tiles (xF, OutF)
constexpr int SQ = 36;   // f32 row stride for 32-wide qh/kh

// ---- WMMA fragment loaders (v_wmma_f32_16x16x32_f16 layouts, wave32) ----
// A fragment (16x32 f16): lane r=lane&15 is row, g=lane>>4; halves j -> k = j + 16g
__device__ __forceinline__ v16h ldA_lds(const _Float16* b, int ld, int r0, int k0) {
  const int lane = threadIdx.x & 31, g = lane >> 4, r = lane & 15;
  const _Float16* p = b + (r0 + r) * ld + k0 + g * 16;
  v16h f;
#pragma unroll
  for (int j = 0; j < 8; ++j) f[j] = p[j];
#pragma unroll
  for (int j = 0; j < 8; ++j) f[8 + j] = p[8 + j];
  return f;
}
// B fragment (32x16 f16): lane c=lane&15 is column, halves j -> k = j + 16g.
// "row-major" = computing X*Y^T with Y stored row-major [col][k] (global f32 source)
__device__ __forceinline__ v16h ldB_rm_g(const float* b, int ld, int c0, int k0) {
  const int lane = threadIdx.x & 31, g = lane >> 4, c = lane & 15;
  const float* p = b + (size_t)(c0 + c) * ld + k0 + g * 16;
  v16h f;
#pragma unroll
  for (int j = 0; j < 16; ++j) f[j] = (_Float16)p[j];
  return f;
}
__device__ __forceinline__ v16h ldB_rm_lds(const _Float16* b, int ld, int c0, int k0) {
  const int lane = threadIdx.x & 31, g = lane >> 4, c = lane & 15;
  const _Float16* p = b + (c0 + c) * ld + k0 + g * 16;
  v16h f;
#pragma unroll
  for (int j = 0; j < 16; ++j) f[j] = p[j];
  return f;
}
// B fragment from Y stored [k][col] row-major (X*Y, e.g. alpha*V)
__device__ __forceinline__ v16h ldB_cm_lds(const _Float16* b, int ld, int c0, int k0) {
  const int lane = threadIdx.x & 31, g = lane >> 4, c = lane & 15;
  v16h f;
#pragma unroll
  for (int j = 0; j < 16; ++j) f[j] = b[(k0 + g * 16 + j) * ld + c0 + c];
  return f;
}
// C/D (16x16 f32): VGPR r holds row r+8g, col = lane&15
__device__ __forceinline__ void st_f32(float* o, int ld, int r0, int c0, const v8f& a) {
  const int lane = threadIdx.x & 31, g = lane >> 4, c = lane & 15;
#pragma unroll
  for (int r = 0; r < 8; ++r) o[(r0 + r + 8 * g) * ld + c0 + c] = a[r];
}
__device__ __forceinline__ void st_f16(_Float16* o, int ld, int r0, int c0, const v8f& a) {
  const int lane = threadIdx.x & 31, g = lane >> 4, c = lane & 15;
#pragma unroll
  for (int r = 0; r < 8; ++r) o[(r0 + r + 8 * g) * ld + c0 + c] = (_Float16)a[r];
}

#define WMMA(acc, a, b) \
  acc = __builtin_amdgcn_wmma_f32_16x16x32_f16(false, (a), false, (b), (short)0, (acc), false, false)

// LDS arena with lifetime-based aliasing (~137 KB of 320 KB/WGP -> 2 WGs/WGP)
constexpr int OFF_Z   = 0;                    // 64*136 f16  (17408 B)  [SpH reuses]
constexpr int OFF_Q   = OFF_Z + 17408;        // 64*136 f16             [AlH reuses]
constexpr int OFF_K   = OFF_Q + 17408;
constexpr int OFF_V   = OFF_K + 17408;
constexpr int OFF_LG  = OFF_V + 17408;        // 64*68 f32 (17408 B)    [OutF reuses LG+QH+KH]
constexpr int OFF_QH  = OFF_LG + 17408;       // 64*36 f32 (9216 B)
constexpr int OFF_KH  = OFF_QH + 9216;        // 64*36 f32 (9216 B)
constexpr int OFF_SM  = OFF_KH + 9216;        // small weights: 128+32+32 f32 (768 B)
constexpr int OFF_XF  = OFF_SM + 768;         // 64*132 f32 x tile (33792 B), persistent
constexpr int SMEM_SZ = OFF_XF + 33792;       // 140032 B

__global__ __launch_bounds__(256) void graph_attn_kernel(
    const float* __restrict__ x, const float* __restrict__ ef, const float* __restrict__ ap,
    const unsigned char* __restrict__ mask, const float* __restrict__ Wq,
    const float* __restrict__ Wk, const float* __restrict__ Wv, const float* __restrict__ e1w,
    const float* __restrict__ e1b, const float* __restrict__ e2w, const float* __restrict__ e2b,
    const float* __restrict__ priorw, const float* __restrict__ theta,
    const float* __restrict__ lng, const float* __restrict__ lnb,
    const float* __restrict__ physw, const float* __restrict__ priw, float* __restrict__ out) {
  __shared__ __align__(16) unsigned char smem[SMEM_SZ];
  _Float16* Zh  = (_Float16*)(smem + OFF_Z);
  _Float16* Qh  = (_Float16*)(smem + OFF_Q);
  _Float16* Kh  = (_Float16*)(smem + OFF_K);
  _Float16* Vh  = (_Float16*)(smem + OFF_V);
  _Float16* SpH = Zh;   // spatial reuses Z region (Z dead after QKV)
  _Float16* AlH = Qh;   // alpha reuses Q region (Q dead after content/qh)
  float* Lg   = (float*)(smem + OFF_LG);
  float* OutF = (float*)(smem + OFF_LG);  // OutF reuses Lg+qhF+khF (dead by phase 7)
  float* qhF  = (float*)(smem + OFF_QH);
  float* khF  = (float*)(smem + OFF_KH);
  float* sWE  = (float*)(smem + OFF_SM);  // 32x4
  float* sB1  = sWE + 128;                // 32
  float* sE2  = sB1 + 32;                 // 32
  float* xF   = (float*)(smem + OFF_XF);  // 64x132 f32, persistent x tile

  const int tid  = threadIdx.x;
  const int wave = tid >> 5;
  const int b    = blockIdx.x;  // 0..127 (= bi*T + t)
  const int bi   = b >> 6;
  const int t    = b & 63;

  // ---- stage small edge-MLP weights ----
  if (tid < 128) sWE[tid] = e1w[(tid >> 2) * 260 + 256 + (tid & 3)];
  if (tid < 32) { sB1[tid] = e1b[tid]; sE2[tid] = e2w[tid]; }

  // ---- stage x slice (64 x 128 f32, row stride 8192 elems) into LDS via TDM ----
#if HAVE_TDM
  if (wave == 0) {
    const unsigned long long ga =
        (unsigned long long)(uintptr_t)x + ((size_t)bi * 4096 + t) * 128ull * 4ull;
    const unsigned lds_off = (unsigned)(uintptr_t)(void*)xF;  // flat LDS addr low 32b = offset
    u32x4 g0;
    g0[0] = 1u;                                    // count=1, no gather
    g0[1] = lds_off;                               // lds_addr
    g0[2] = (unsigned)(ga & 0xFFFFFFFFu);          // global_addr[31:0]
    g0[3] = (unsigned)((ga >> 32) & 0x01FFFFFFu) | (2u << 30);  // global_addr[56:32] | type=2
    i32x8 g1;
    g1[0] = (2 << 16) | (1 << 20) | (6 << 22) | (3 << 25);  // 4B elems, pad: +4dw per 128dw
    g1[1] = (128 << 16);  // tensor_dim0 = 128
    g1[2] = (64 << 16);   // tensor_dim1 = 64
    g1[3] = (128 << 16);  // tile_dim0 = 128
    g1[4] = 64;           // tile_dim1 = 64, tile_dim2 = 0
    g1[5] = 8192;         // tensor_dim0_stride = T*D = 8192
    g1[6] = 0;
    g1[7] = 0;
    const i32x4 gz = {0, 0, 0, 0};
#if __clang_major__ >= 23
    const i32x8 g4z = {0, 0, 0, 0, 0, 0, 0, 0};
    __builtin_amdgcn_tensor_load_to_lds(g0, g1, gz, gz, g4z, 0);
#else
    __builtin_amdgcn_tensor_load_to_lds(g0, g1, gz, gz, 0);
#endif
    __builtin_amdgcn_s_wait_tensorcnt(0);
  }
#else
  for (int i = tid; i < 64 * 32; i += 256) {
    const int n = i >> 5, d4 = i & 31;
    const float4 v = ((const float4*)(x + (((size_t)bi * 64 + n) * 64 + t) * 128))[d4];
    ((float4*)(xF + n * SO))[d4] = v;
  }
#endif
  __syncthreads();

  // ---- convert x tile to f16 Z (128-bit LDS reads, 64-bit LDS stores) ----
  for (int i = tid; i < 64 * 32; i += 256) {
    const int n = i >> 5, d4 = i & 31;
    const float4 v = ((const float4*)(xF + n * SO))[d4];
    union { _Float16 h[4]; unsigned long long u; } pk;
    pk.h[0] = (_Float16)v.x; pk.h[1] = (_Float16)v.y;
    pk.h[2] = (_Float16)v.z; pk.h[3] = (_Float16)v.w;
    *(unsigned long long*)(Zh + n * SZ + d4 * 4) = pk.u;
  }
  __syncthreads();

  // ---- phase 2a: Q/K/V = Z * W^T  (96 tile-jobs, 12 per wave) ----
  for (int j = 0; j < 12; ++j) {
    const int job = wave * 12 + j;
    const int matid = job >> 5;          // 0=Q 1=K 2=V
    const int tile = job & 31;
    const int tm = tile >> 3, tn = tile & 7;
    const float* W = (matid == 0) ? Wq : (matid == 1) ? Wk : Wv;
    _Float16* Dst  = (matid == 0) ? Qh : (matid == 1) ? Kh : Vh;
    v8f acc = {};
#pragma unroll
    for (int kc = 0; kc < 4; ++kc) {
      v16h a = ldA_lds(Zh, SZ, tm * 16, kc * 32);
      v16h bf = ldB_rm_g(W, 128, tn * 16, kc * 32);
      WMMA(acc, a, bf);
    }
    st_f16(Dst, SZ, tm * 16, tn * 16, acc);
  }
  __syncthreads();

  // ---- phase 2b: content = Q*K^T (16 tiles) and qh/kh = {Q,K}*w^T (16 tiles) ----
  for (int j = 0; j < 4; ++j) {
    const int job = wave * 4 + j;  // 0..31, wave-uniform
    if (job < 16) {
      const int tm = job >> 2, tn = job & 3;
      v8f acc = {};
#pragma unroll
      for (int kc = 0; kc < 4; ++kc) {
        v16h a = ldA_lds(Qh, SZ, tm * 16, kc * 32);
        v16h bf = ldB_rm_lds(Kh, SZ, tn * 16, kc * 32);
        WMMA(acc, a, bf);
      }
      st_f32(Lg, SL, tm * 16, tn * 16, acc);
    } else {
      const int jj = job - 16;
      const int matid = jj >> 3;           // 0: qh=Q*w_qi^T, 1: kh=K*w_kj^T
      const int tile = jj & 7;
      const int tm = tile >> 1, tn = tile & 1;
      const _Float16* Src = matid ? Kh : Qh;
      float* Dstf = matid ? khF : qhF;
      const float* Wb = e1w + (matid ? 128 : 0);  // ld = 260
      v8f acc = {};
#pragma unroll
      for (int kc = 0; kc < 4; ++kc) {
        v16h a = ldA_lds(Src, SZ, tm * 16, kc * 32);
        v16h bf = ldB_rm_g(Wb, 260, tn * 16, kc * 32);
        WMMA(acc, a, bf);
      }
      st_f32(Dstf, SQ, tm * 16, tn * 16, acc);
    }
  }
  __syncthreads();

  // ---- phase 4: edge MLP + prior + mask -> logits (VALU, 4096 pairs) ----
  {
    const float pw = physw[0], prw = priw[0], eb = e2b[0];
    const float p0 = priorw[0], p1 = priorw[1], p2 = priorw[2], p3 = priorw[3], p4 = priorw[4];
    const float inv_sqrt_d = 0.0883883476483184f;  // 1/sqrt(128)
    for (int p = tid; p < 64 * 64; p += 256) {
      const int n = p >> 6, m = p & 63;
      const float4 e = *(const float4*)(ef + (((size_t)b * 64 + n) * 64 + m) * 4);
      float phys = 0.f;
#pragma unroll
      for (int h = 0; h < 32; ++h) {
        float v = qhF[n * SQ + h] + khF[m * SQ + h] + sB1[h];
        v = fmaf(sWE[h * 4 + 0], e.x, v);
        v = fmaf(sWE[h * 4 + 1], e.y, v);
        v = fmaf(sWE[h * 4 + 2], e.z, v);
        v = fmaf(sWE[h * 4 + 3], e.w, v);
        phys = fmaf(fmaxf(v, 0.f), sE2[h], phys);
      }
      phys += eb;
      const float* app = ap + (((size_t)b * 64 + n) * 64 + m) * 5;
      float a0 = app[0] * p0 + app[1] * p1 + app[2] * p2 + app[3] * p3 + app[4] * p4;
      if (!__builtin_isfinite(a0)) a0 = 0.f;
      a0 = fmaxf(a0, 0.f);
      float lg = Lg[n * SL + m] * inv_sqrt_d + pw * phys + prw * __logf(a0 + 1e-6f);
      const bool inval = (mask[bi * 64 + n] | mask[bi * 64 + m]) != 0;
      Lg[n * SL + m] = inval ? -1.0e9f : lg;
    }
  }
  __syncthreads();

  // ---- phase 5: row softmax -> alpha (f16) ----
  if (tid < 64) {
    const int n = tid;
    float mx = -3.4e38f;
    for (int m = 0; m < 64; ++m) mx = fmaxf(mx, Lg[n * SL + m]);
    float s = 0.f;
    for (int m = 0; m < 64; ++m) {
      const float e = __expf(Lg[n * SL + m] - mx);
      Lg[n * SL + m] = e;
      s += e;
    }
    const float inv = 1.f / s;
    for (int m = 0; m < 64; ++m) AlH[n * SA + m] = (_Float16)(Lg[n * SL + m] * inv);
  }
  __syncthreads();

  // ---- phase 6: spatial = alpha * V  (32 tiles, K=64) ----
  for (int j = 0; j < 4; ++j) {
    const int tile = wave * 4 + j;
    const int tm = tile >> 3, tn = tile & 7;
    v8f acc = {};
#pragma unroll
    for (int kc = 0; kc < 2; ++kc) {
      v16h a = ldA_lds(AlH, SA, tm * 16, kc * 32);
      v16h bf = ldB_cm_lds(Vh, SZ, tn * 16, kc * 32);
      WMMA(acc, a, bf);
    }
    st_f16(SpH, SZ, tm * 16, tn * 16, acc);
  }
  __syncthreads();

  // ---- phase 7: outp = spatial * theta^T (32 tiles, K=128) ----
  for (int j = 0; j < 4; ++j) {
    const int tile = wave * 4 + j;
    const int tm = tile >> 3, tn = tile & 7;
    v8f acc = {};
#pragma unroll
    for (int kc = 0; kc < 4; ++kc) {
      v16h a = ldA_lds(SpH, SZ, tm * 16, kc * 32);
      v16h bf = ldB_rm_g(theta, 128, tn * 16, kc * 32);
      WMMA(acc, a, bf);
    }
    st_f32(OutF, SO, tm * 16, tn * 16, acc);
  }
  __syncthreads();

  // ---- phase 8: residual (x from LDS) + LayerNorm + transposed, masked store ----
  if (tid < 64) {
    const int n = tid;
    const float4* xr  = (const float4*)(xF + n * SO);
    const float4* orw = (const float4*)(OutF + n * SO);
    float s = 0.f, s2 = 0.f;
    for (int d4 = 0; d4 < 32; ++d4) {
      const float4 a = xr[d4], c = orw[d4];
      const float r0 = a.x + c.x, r1 = a.y + c.y, r2 = a.z + c.z, r3 = a.w + c.w;
      s += r0 + r1 + r2 + r3;
      s2 += r0 * r0 + r1 * r1 + r2 * r2 + r3 * r3;
    }
    const float mu = s * (1.f / 128.f);
    const float var = fmaxf(s2 * (1.f / 128.f) - mu * mu, 0.f);
    const float rinv = rsqrtf(var + 1e-5f);
    float4* orow = (float4*)(out + (((size_t)bi * 64 + n) * 64 + t) * 128);
    const bool msk = mask[bi * 64 + n] != 0;
    for (int d4 = 0; d4 < 32; ++d4) {
      const float4 a = xr[d4], c = orw[d4];
      const float4 g = ((const float4*)lng)[d4], bb = ((const float4*)lnb)[d4];
      float4 o;
      o.x = (a.x + c.x - mu) * rinv * g.x + bb.x;
      o.y = (a.y + c.y - mu) * rinv * g.y + bb.y;
      o.z = (a.z + c.z - mu) * rinv * g.z + bb.z;
      o.w = (a.w + c.w - mu) * rinv * g.w + bb.w;
      if (msk) { o.x = 0.f; o.y = 0.f; o.z = 0.f; o.w = 0.f; }
      orow[d4] = o;
    }
  }
}

extern "C" void kernel_launch(void* const* d_in, const int* in_sizes, int n_in,
                              void* d_out, int out_size, void* d_ws, size_t ws_size,
                              hipStream_t stream) {
  (void)in_sizes; (void)n_in; (void)d_ws; (void)ws_size; (void)out_size;
  graph_attn_kernel<<<128, 256, 0, stream>>>(
      (const float*)d_in[0],          // x
      (const float*)d_in[1],          // edge_features
      (const float*)d_in[2],          // A_prior
      (const unsigned char*)d_in[3],  // entity_padding_mask (bool)
      (const float*)d_in[4],          // Wq
      (const float*)d_in[5],          // Wk
      (const float*)d_in[6],          // Wv
      (const float*)d_in[7],          // e1_w
      (const float*)d_in[8],          // e1_b
      (const float*)d_in[9],          // e2_w
      (const float*)d_in[10],         // e2_b
      (const float*)d_in[11],         // prior_w
      (const float*)d_in[12],         // theta_w
      (const float*)d_in[13],         // ln_g
      (const float*)d_in[14],         // ln_b
      (const float*)d_in[15],         // phys_weight
      (const float*)d_in[16],         // prior_weight
      (float*)d_out);
}
